// LSTM_4372276707354
// MI455X (gfx1250) — matrix operand order repaired and checked by reading, not compiled, not run on hardware
//
#include <hip/hip_runtime.h>

typedef __bf16 bf16;
typedef __attribute__((ext_vector_type(16))) bf16  v16bf;
typedef __attribute__((ext_vector_type(8)))  bf16  v8bf;
typedef __attribute__((ext_vector_type(8)))  float v8f;
typedef __attribute__((ext_vector_type(4)))  unsigned int u32x4;
typedef __attribute__((ext_vector_type(8)))  int i32x8;
typedef __attribute__((ext_vector_type(4)))  int i32x4;

#define Bsz 64
#define Ssz 1024
#define Isz 512
#define Hsz 512

// ---------------- pack x: [B][S][I] f32 -> [S][B][I] bf16 ----------------
__global__ __launch_bounds__(256) void pack_x_kernel(const float* __restrict__ x,
                                                     bf16* __restrict__ xT) {
  size_t idx = (size_t)blockIdx.x * 256 + threadIdx.x;   // over S*B*I
  int i = (int)(idx % Isz);
  size_t sb = idx / Isz;
  int b = (int)(sb % Bsz);
  int s = (int)(sb / Bsz);
  xT[idx] = (bf16)x[((size_t)b * Ssz + s) * Isz + i];
}

// ------------- pack weights: 8x [K=512][N=512] f32 row-major -------------
// -> Wt[g*2+uv][n][k] bf16 (column-major per matrix, contiguous K per column)
struct WPtrs { const float* p[8]; };

__global__ __launch_bounds__(256) void pack_w_kernel(WPtrs wp, bf16* __restrict__ Wt) {
  int idx = blockIdx.x * 256 + threadIdx.x;              // over 8*512*512
  int k = idx & 511;
  int n = (idx >> 9) & 511;
  int m = idx >> 18;
  Wt[idx] = (bf16)wp.p[m][(size_t)k * Hsz + n];
}

// ---------------- zero initial h (bf16) and c (f32) ----------------
__global__ __launch_bounds__(256) void init_state_kernel(bf16* __restrict__ h0,
                                                         float* __restrict__ c) {
  int idx = blockIdx.x * 256 + threadIdx.x;
  h0[idx] = (bf16)0.0f;
  c[idx]  = 0.0f;
}

// ---------------- one LSTM timestep, fully fused ----------------
// grid = (B/16)*(H/16) = 128 blocks, 512 threads (16 waves).
// wave w = q*4 + g : gate g in 0..3, K-quarter q in 0..3
//   q=0: x[:,0:256)   q=1: x[:,256:512)   q=2: h[:,0:256)   q=3: h[:,256:512)
// A slabs (16 rows x 512 K each for x and h) staged to LDS by the TDM.
__global__ __launch_bounds__(512) void lstm_step_kernel(
    const bf16* __restrict__ xT_t,   // [64][512] bf16 (this timestep)
    const bf16* __restrict__ Wt,     // [8][512][512] bf16, (g*2+uv, n, k)
    const float* __restrict__ bF, const float* __restrict__ bI,
    const float* __restrict__ bC, const float* __restrict__ bO,
    const bf16* __restrict__ h_in,   // [64][512] bf16
    bf16* __restrict__ h_out,        // [64][512] bf16
    float* __restrict__ c_state,     // [64][512] f32
    float* __restrict__ out,         // [B][S][H] f32
    int t)
{
  __shared__ bf16  sA[2][16][512];   // [0]=x slab, [1]=h slab : 32 KB
  __shared__ float gl[16][16][16];   // per-wave partial gate tiles: 16 KB

  const int tid   = threadIdx.x;
  const int w     = tid >> 5;        // wave 0..15
  const int lane  = tid & 31;
  const int g     = w & 3;           // gate: 0=f 1=i 2=c 3=o
  const int q     = w >> 2;          // K-quarter
  const int slab  = q >> 1;          // 0=x (U weights), 1=h (V weights)
  const int kq    = (q & 1) * 256;   // K offset within slab
  const int rt    = blockIdx.x & 3;  // b-tile (B/16 = 4)
  const int ct    = blockIdx.x >> 2; // h col-tile (H/16 = 32)
  const int l15   = lane & 15;       // A row in tile / B col in tile
  const int khalf = lane >> 4;       // 0 or 1

  const int nCol = ct * 16 + l15;    // global col this lane's B frag covers

  // ---- TDM: DMA both A slabs (16 rows x 512 bf16 each) into LDS ----
  if (tid < 32) {                    // wave 0 only (s_cbranch_execz skips rest)
    unsigned long long gax =
        (unsigned long long)(uintptr_t)(xT_t + (size_t)rt * 16 * Isz);
    unsigned long long gah =
        (unsigned long long)(uintptr_t)(h_in + (size_t)rt * 16 * Hsz);
    unsigned int ldsx = (unsigned int)(uintptr_t)(&sA[0][0][0]);
    unsigned int ldsh = (unsigned int)(uintptr_t)(&sA[1][0][0]);
    // D# group0: count=1 | lds_addr | global_addr[56:0] | type=2
    u32x4 g0x = {1u, ldsx, (unsigned int)gax,
                 (unsigned int)(gax >> 32) | (2u << 30)};
    u32x4 g0h = {1u, ldsh, (unsigned int)gah,
                 (unsigned int)(gah >> 32) | (2u << 30)};
    // D# group1: data_size=2B; tensor_dim0=512, tensor_dim1=16;
    //            tile_dim0=512, tile_dim1=16, tile_dim2=0; dim0_stride=512
    i32x8 g1v = {0x00010000, 0x02000000, 0x00100000, 0x02000000,
                 16, 512, 0, 0};
    i32x4 z4 = {0, 0, 0, 0};
    i32x8 z8 = {0, 0, 0, 0, 0, 0, 0, 0};
    __builtin_amdgcn_tensor_load_to_lds(g0x, g1v, z4, z4, z8, 0);
    __builtin_amdgcn_tensor_load_to_lds(g0h, g1v, z4, z4, z8, 0);
    __builtin_amdgcn_s_wait_tensorcnt(0);
  }
  __syncthreads();

  // ---- 8-deep WMMA chain over this wave's K quarter ----
  const bf16* Bcol = Wt + ((size_t)(g * 2 + slab) * Hsz + nCol) * 512 + kq;
  v8f acc = {};
#pragma unroll
  for (int k0 = 0; k0 < 256; k0 += 32) {
    // A 16x32 bf16 fragment from LDS: two contiguous 16B chunks per lane
    const bf16* Ap = &sA[slab][l15][kq + k0 + khalf * 16];
    union { v16bf v; v8bf h[2]; } ua;
    ua.h[0] = *(const v8bf*)(Ap);
    ua.h[1] = *(const v8bf*)(Ap + 8);
    // B 32x16 bf16 fragment from L2-resident packed weights: one 32B chunk
    v16bf bfrag = *(const v16bf*)(Bcol + k0 + khalf * 16);
    acc = __builtin_amdgcn_wmma_f32_16x16x32_bf16(
            /*neg_a=*/false, ua.v, /*neg_b=*/false, bfrag,
            /*c_mod=*/(short)0, acc, /*reuse_a=*/false, /*reuse_b=*/false);
  }

  // ---- partial tiles -> LDS (C layout: lane holds col l15, rows khalf*8+v) --
#pragma unroll
  for (int v = 0; v < 8; ++v)
    gl[w][khalf * 8 + v][l15] = acc[v];
  __syncthreads();

  // ---- reduce K-quarters, apply LSTM cell, write state + output ----
  if (tid < 256) {
    int m = tid >> 4, n = tid & 15;
    int b   = rt * 16 + m;
    int col = ct * 16 + n;
    float gf = gl[0][m][n] + gl[4][m][n] + gl[8][m][n] + gl[12][m][n] + bF[col];
    float gi = gl[1][m][n] + gl[5][m][n] + gl[9][m][n] + gl[13][m][n] + bI[col];
    float gc = gl[2][m][n] + gl[6][m][n] + gl[10][m][n] + gl[14][m][n] + bC[col];
    float go = gl[3][m][n] + gl[7][m][n] + gl[11][m][n] + gl[15][m][n] + bO[col];
    float f  = 1.0f / (1.0f + __expf(-gf));
    float ii = 1.0f / (1.0f + __expf(-gi));
    float cb = tanhf(gc);
    float o  = 1.0f / (1.0f + __expf(-go));
    size_t sidx = (size_t)b * Hsz + col;
    float cn = f * c_state[sidx] + ii * cb;
    float hn = o * tanhf(cn);
    c_state[sidx] = cn;
    h_out[sidx]   = (bf16)hn;
    out[((size_t)b * Ssz + t) * Hsz + col] = hn;
  }
}

// ---------------- host launcher ----------------
extern "C" void kernel_launch(void* const* d_in, const int* in_sizes, int n_in,
                              void* d_out, int out_size, void* d_ws, size_t ws_size,
                              hipStream_t stream) {
  const float* x = (const float*)d_in[0];
  WPtrs wp;
  wp.p[0] = (const float*)d_in[1];   // Uf
  wp.p[1] = (const float*)d_in[2];   // Vf
  wp.p[2] = (const float*)d_in[4];   // Ui
  wp.p[3] = (const float*)d_in[5];   // Vi
  wp.p[4] = (const float*)d_in[7];   // Uc
  wp.p[5] = (const float*)d_in[8];   // Vc
  wp.p[6] = (const float*)d_in[10];  // Uo
  wp.p[7] = (const float*)d_in[11];  // Vo
  const float* bF = (const float*)d_in[3];
  const float* bI = (const float*)d_in[6];
  const float* bC = (const float*)d_in[9];
  const float* bO = (const float*)d_in[12];

  // workspace layout (all 256B-aligned)
  char* ws = (char*)d_ws;
  bf16*  xT  = (bf16*)ws;                                  // 64 MB
  bf16*  Wt  = (bf16*)(ws + 67108864);                     // 4 MB
  bf16*  h0  = (bf16*)(ws + 67108864 + 4194304);           // 64 KB
  bf16*  h1  = h0 + (size_t)Bsz * Hsz;                     // 64 KB
  float* cst = (float*)(ws + 67108864 + 4194304 + 131072); // 128 KB
  float* out = (float*)d_out;

  pack_x_kernel<<<(Ssz * Bsz * Isz) / 256, 256, 0, stream>>>(x, xT);
  pack_w_kernel<<<(8 * 512 * 512) / 256, 256, 0, stream>>>(wp, Wt);
  init_state_kernel<<<(Bsz * Hsz) / 256, 256, 0, stream>>>(h0, cst);

  for (int t = 0; t < Ssz; ++t) {
    const bf16* hin = (t & 1) ? h1 : h0;
    bf16* hout      = (t & 1) ? h0 : h1;
    lstm_step_kernel<<<dim3((Bsz / 16) * (Hsz / 16)), dim3(512), 0, stream>>>(
        xT + (size_t)t * Bsz * Isz, Wt, bF, bI, bC, bO, hin, hout, cst, out, t);
  }
}
